// RNNModel_24550033063947
// MI455X (gfx1250) — matrix operand order repaired and checked by reading, not compiled, not run on hardware
//
#include <hip/hip_runtime.h>

// ---------------------------------------------------------------------------
// Problem constants (fixed by the reference): B=256, T=120, E=256, H=1024, C=5
// LSTM gate order (TF LSTMCell): i, j, f, o ; forget bias = 1.0
// ---------------------------------------------------------------------------
constexpr int B_   = 256;
constexpr int T_   = 120;
constexpr int E_   = 256;
constexpr int H_   = 1024;
constexpr int C_   = 5;
constexpr int KTOT = E_ + H_;   // 1280
constexpr int N4H  = 4 * H_;    // 4096

typedef __attribute__((ext_vector_type(16))) __bf16       v16bf;
typedef __attribute__((ext_vector_type(8)))  float        v8f;
typedef __attribute__((ext_vector_type(4)))  unsigned int u32x4;

union Frag {
    struct { u32x4 lo, hi; } u;   // two global_load_b128 per fragment
    v16bf v;
};

// One pipeline stage: A fragment + 4 gate B fragments (10 b128 loads).
struct Stage {
    Frag a, b0, b1, b2, b3;
};

// bf16 <-> f32 helpers (round-to-nearest-even), no libdevice dependency.
static __device__ __forceinline__ __bf16 f2bf(float f) {
    unsigned u = __builtin_bit_cast(unsigned, f);
    unsigned r = (u + 0x7fffu + ((u >> 16) & 1u)) >> 16;
    unsigned short s = (unsigned short)r;
    return __builtin_bit_cast(__bf16, s);
}
static __device__ __forceinline__ float bf2f(__bf16 b) {
    unsigned short s = __builtin_bit_cast(unsigned short, b);
    unsigned u = ((unsigned)s) << 16;
    return __builtin_bit_cast(float, u);
}
static __device__ __forceinline__ float sigmoidf_(float x) {
    return 1.0f / (1.0f + __expf(-x));
}

// ---------------------------------------------------------------------------
// Kernel 1: embedding gather + f32 -> bf16.  x_bf16[(b*T+t)*E + e]
// ---------------------------------------------------------------------------
__global__ __launch_bounds__(256) void embed_bf16_kernel(
    const int* __restrict__ tok, const float* __restrict__ emb,
    __bf16* __restrict__ xb)
{
    size_t i  = (size_t)blockIdx.x * 256 + threadIdx.x;   // < B*T*E
    int    e  = (int)(i & (E_ - 1));
    size_t bt = i >> 8;                                    // E_ == 256
    int    v  = tok[bt];
    xb[i] = f2bf(emb[(size_t)v * E_ + e]);
}

// ---------------------------------------------------------------------------
// Kernel 2: W (KTOT x 4H, row-major f32) -> Wt (4H x KTOT, bf16)
// N-major so each lane's WMMA B-fragment is 32 contiguous bytes.
// ---------------------------------------------------------------------------
__global__ __launch_bounds__(256) void wtranspose_bf16_kernel(
    const float* __restrict__ W, __bf16* __restrict__ Wt)
{
    size_t i = (size_t)blockIdx.x * 256 + threadIdx.x;    // < 4H*KTOT
    int k = (int)(i % KTOT);
    int n = (int)(i / KTOT);
    Wt[i] = f2bf(W[(size_t)k * N4H + n]);
}

// ---------------------------------------------------------------------------
// Kernel 3: zero h0 (bf16) and c (f32).  B*H = 262144 elements each.
// ---------------------------------------------------------------------------
__global__ __launch_bounds__(256) void zero_state_kernel(
    __bf16* __restrict__ h0, float* __restrict__ c)
{
    int i = blockIdx.x * 256 + threadIdx.x;
    h0[i] = f2bf(0.0f);
    c[i]  = 0.0f;
}

// ---------------------------------------------------------------------------
// Kernel 4: one LSTM timestep, fully fused, software-pipelined.
//   z = [x_t | h_prev] @ W + b   (K = 1280 = 8 x-chunks + 32 h-chunks of 32)
//   c_new = sigmoid(f+1)*c + sigmoid(i)*tanh(j) ; h = sigmoid(o)*tanh(c_new)
// One wave => one 16(M=batch) x 16(N=h) tile, 4 gate accumulators sharing A.
// Double-buffered: step s+1's 10 loads are issued before step s's 4 WMMAs,
// so the wait before the WMMA burst is partial, not s_wait_loadcnt 0.
// Grid: 16 M-tiles * 16 N-groups = 256 blocks of 4 waves (128 threads).
// ---------------------------------------------------------------------------
__global__ __launch_bounds__(128) void lstm_cell_kernel(
    const __bf16* __restrict__ xb,     // (B, T, E) bf16
    const __bf16* __restrict__ hprev,  // (B, H) bf16
    __bf16*       __restrict__ hout,   // (B, H) bf16
    float*        __restrict__ cst,    // (B, H) f32, updated in place
    const __bf16* __restrict__ Wt,     // (4H, KTOT) bf16, N-major
    const float*  __restrict__ bias,   // (4H,) f32
    int t)
{
    const int  lane   = threadIdx.x & 31;
    const int  wave   = threadIdx.x >> 5;
    const int  tile_m = blockIdx.x >> 4;                 // 0..15
    const int  tile_n = ((blockIdx.x & 15) << 2) | wave; // 0..63
    const int  m_base = tile_m * 16;
    const int  n_base = tile_n * 16;
    const bool hiHalf = lane >= 16;
    const int  arow   = lane & 15;          // A-matrix M index for this lane
    const int  ncol   = n_base + (lane & 15);

    // A-operand source rows: K 0..255 from x_t, K 256..1279 from h_prev
    const __bf16* xrow = xb    + ((size_t)(m_base + arow) * T_ + t) * E_;
    const __bf16* hrow = hprev +  (size_t)(m_base + arow) * H_;

    // B-operand rows (one row per lane column) for the four gates
    const __bf16* w0 = Wt + (size_t)(0 * H_ + ncol) * KTOT;
    const __bf16* w1 = Wt + (size_t)(1 * H_ + ncol) * KTOT;
    const __bf16* w2 = Wt + (size_t)(2 * H_ + ncol) * KTOT;
    const __bf16* w3 = Wt + (size_t)(3 * H_ + ncol) * KTOT;

    v8f acc_i = {}, acc_j = {}, acc_f = {}, acc_o = {};

    // Fragment k order used by this code (16-bit, contiguous; A and B agree):
    //   A 16x32: lane<16 K=k0+0..15 ; lane>=16 K=k0+16..31 (contiguous, matches B)
    //   B 32x16: lane<16 K=k0+0..15 ; lane>=16 K=k0+16..31 (contiguous 32B)
    const int aOff = hiHalf ? 16 : 0;
    const int bOff = hiHalf ? 16 : 0;

    auto load_stage = [&](Stage& d, const __bf16* abase, int relk, int k0w) {
        const int kA = relk + aOff;
        d.a.u.lo = *(const u32x4*)(abase + kA);
        d.a.u.hi = *(const u32x4*)(abase + kA + 8);
        const int kB = k0w + bOff;
        d.b0.u.lo = *(const u32x4*)(w0 + kB); d.b0.u.hi = *(const u32x4*)(w0 + kB + 8);
        d.b1.u.lo = *(const u32x4*)(w1 + kB); d.b1.u.hi = *(const u32x4*)(w1 + kB + 8);
        d.b2.u.lo = *(const u32x4*)(w2 + kB); d.b2.u.hi = *(const u32x4*)(w2 + kB + 8);
        d.b3.u.lo = *(const u32x4*)(w3 + kB); d.b3.u.hi = *(const u32x4*)(w3 + kB + 8);
        // Prefetch two steps ahead on the gate-0 row (global_prefetch_b8)
        __builtin_prefetch(w0 + kB + 64, 0, 1);
    };
    auto compute_stage = [&](Stage& d) {
        acc_i = __builtin_amdgcn_wmma_f32_16x16x32_bf16(false, d.a.v, false, d.b0.v,
                                                        (short)0, acc_i, false, false);
        acc_j = __builtin_amdgcn_wmma_f32_16x16x32_bf16(false, d.a.v, false, d.b1.v,
                                                        (short)0, acc_j, false, false);
        acc_f = __builtin_amdgcn_wmma_f32_16x16x32_bf16(false, d.a.v, false, d.b2.v,
                                                        (short)0, acc_f, false, false);
        acc_o = __builtin_amdgcn_wmma_f32_16x16x32_bf16(false, d.a.v, false, d.b3.v,
                                                        (short)0, acc_o, false, false);
    };

    Stage buf0, buf1;

    // ---- x phase: steps 0..7 (K = 0..255 from xrow) ----
    load_stage(buf0, xrow, 0, 0);
    for (int sp = 0; sp < 8; sp += 2) {
        load_stage(buf1, xrow, (sp + 1) * 32, (sp + 1) * 32);
        compute_stage(buf0);
        if (sp + 2 < 8) load_stage(buf0, xrow, (sp + 2) * 32, (sp + 2) * 32);
        else            load_stage(buf0, hrow, 0, E_);       // bridge to h phase
        compute_stage(buf1);
    }

    // ---- h phase: steps 8..39 (K = 256..1279 from hrow) ----
    for (int sp = 8; sp < 40; sp += 2) {
        load_stage(buf1, hrow, (sp - 7) * 32, (sp + 1) * 32);
        compute_stage(buf0);
        if (sp + 2 < 40) load_stage(buf0, hrow, (sp - 6) * 32, (sp + 2) * 32);
        compute_stage(buf1);
    }

    // ---- fused gate epilogue ----
    const float bi = bias[0 * H_ + ncol];
    const float bj = bias[1 * H_ + ncol];
    const float bf = bias[2 * H_ + ncol];
    const float bo = bias[3 * H_ + ncol];

    const int rbase = hiHalf ? 8 : 0;   // C/D layout: M = r + 8*(lane>=16)
#pragma unroll
    for (int r = 0; r < 8; ++r) {
        const int brow = m_base + rbase + r;
        const size_t idx = (size_t)brow * H_ + ncol;
        const float zi = acc_i[r] + bi;
        const float zj = acc_j[r] + bj;
        const float zf = acc_f[r] + bf;
        const float zo = acc_o[r] + bo;
        const float c_old = cst[idx];
        const float c_new = sigmoidf_(zf + 1.0f) * c_old
                          + sigmoidf_(zi) * tanhf(zj);
        const float h_new = sigmoidf_(zo) * tanhf(c_new);
        cst[idx]  = c_new;
        hout[idx] = f2bf(h_new);
    }
}

// ---------------------------------------------------------------------------
// Kernel 5: output projection for one timestep: preds[b,t,c] = h . U[:,c] + b2
// One wave per (b, c) pair: 256*5 = 1280 waves = 320 blocks of 128 threads.
// ---------------------------------------------------------------------------
__global__ __launch_bounds__(128) void proj_kernel(
    const __bf16* __restrict__ h,   // (B, H) bf16
    const float*  __restrict__ U,   // (H, C) f32
    const float*  __restrict__ b2,  // (C,)  f32
    float*        __restrict__ out, // (B, T, C) f32
    int t)
{
    const int gid  = blockIdx.x * 128 + threadIdx.x;
    const int wave = gid >> 5;
    const int lane = gid & 31;
    const int b    = wave / C_;
    const int cc   = wave % C_;

    float s = 0.0f;
    for (int k = lane; k < H_; k += 32)
        s += bf2f(h[(size_t)b * H_ + k]) * U[(size_t)k * C_ + cc];
#pragma unroll
    for (int off = 16; off > 0; off >>= 1)
        s += __shfl_down(s, off, 32);
    if (lane == 0)
        out[((size_t)b * T_ + t) * C_ + cc] = s + b2[cc];
}

// ---------------------------------------------------------------------------
// Host side: one-time conversions, then 120 x (cell + projection) on stream.
// Workspace use: ~28.3 MB.
// ---------------------------------------------------------------------------
extern "C" void kernel_launch(void* const* d_in, const int* in_sizes, int n_in,
                              void* d_out, int out_size, void* d_ws, size_t ws_size,
                              hipStream_t stream) {
    (void)in_sizes; (void)n_in; (void)out_size; (void)ws_size;

    const int*   tok = (const int*)  d_in[0];  // (B, T, 1) int32
    const float* emb = (const float*)d_in[1];  // (V, E)
    const float* W   = (const float*)d_in[2];  // (E+H, 4H)
    const float* bl  = (const float*)d_in[3];  // (4H,)
    const float* U   = (const float*)d_in[4];  // (H, C)
    const float* b2  = (const float*)d_in[5];  // (C,)
    float* out = (float*)d_out;                // (B, T, C)

    char*  ws  = (char*)d_ws;
    size_t off = 0;
    auto take = [&](size_t bytes) -> char* {
        char* p = ws + off;
        off = (off + bytes + 255) & ~(size_t)255;
        return p;
    };
    __bf16* xb  = (__bf16*)take((size_t)B_ * T_ * E_ * 2);   // 15.7 MB
    __bf16* Wt  = (__bf16*)take((size_t)N4H * KTOT * 2);     // 10.5 MB
    __bf16* h0  = (__bf16*)take((size_t)B_ * H_ * 2);        // 0.5 MB
    __bf16* h1  = (__bf16*)take((size_t)B_ * H_ * 2);        // 0.5 MB
    float*  cst = (float*) take((size_t)B_ * H_ * 4);        // 1.0 MB

    // One-time prep
    embed_bf16_kernel<<<dim3((B_ * T_ * E_) / 256), dim3(256), 0, stream>>>(tok, emb, xb);
    wtranspose_bf16_kernel<<<dim3((N4H * KTOT) / 256), dim3(256), 0, stream>>>(W, Wt);
    zero_state_kernel<<<dim3((B_ * H_) / 256), dim3(256), 0, stream>>>(h0, cst);

    // Sequential scan over T with ping-pong h buffers
    __bf16* hb[2] = { h0, h1 };
    for (int t = 0; t < T_; ++t) {
        __bf16* hin  = hb[t & 1];
        __bf16* hout = hb[(t + 1) & 1];
        lstm_cell_kernel<<<dim3(256), dim3(128), 0, stream>>>(
            xb, hin, hout, cst, Wt, bl, t);
        proj_kernel<<<dim3(320), dim3(128), 0, stream>>>(hout, U, b2, out, t);
    }
}
